// ArcFaceLossAdaptiveMarginSubcenter_25340307046404
// MI455X (gfx1250) — matrix operand order repaired and checked by reading, not compiled, not run on hardware
//
#include <hip/hip_runtime.h>
#include <hip/hip_bf16.h>

// ---------------------------------------------------------------------------
// ArcFace adaptive-margin subcenter loss for MI455X (gfx1250, wave32, WMMA)
//
// B=512, D=512, C=100000, K=3, S=30
//  1) normalize features -> f16  fn[512][512]            (ws)
//  2) normalize weights, permute subcenter-major + zero pad:
//       wn[k*CPAD + c][d] = weight[c*3+k][d]/||.||, CPAD=100096 (=391*256)
//  3) WMMA f16 GEMM (f32 accum). A block-tile (64x512 f16 = 64 KB) is staged
//     once into LDS via global_load_async_to_lds_b128; per subcenter block
//     the full K=512 reduction runs and a per-lane register max fuses the
//     subcenter reduction. Store S*cosine.
//  4) tiny epilogue applies the adaptive margin at (row, label)
// ---------------------------------------------------------------------------

#define BB   512
#define DD   512
#define CC   100000
#define KSUB 3
#define SS   30.0f
#define CPAD 100096            // 391 * 256, multiple of block-N
#define LDS_PITCH 520          // 512 + 8 halves: rows shift 16B across banks

typedef __attribute__((ext_vector_type(16))) _Float16 v16h;
typedef __attribute__((ext_vector_type(8)))  _Float16 f16x8;
typedef __attribute__((ext_vector_type(8)))  float    v8f;
typedef __attribute__((ext_vector_type(4)))  float    f32x4;

// ---------------------------------------------------------------------------
// Normalize feature rows (f32 -> unit-norm f16). One wave per row.
// ---------------------------------------------------------------------------
__global__ __launch_bounds__(256)
void arcface_norm_features(const float* __restrict__ feat, _Float16* __restrict__ fn)
{
    const int lane = threadIdx.x & 31;
    const int wave = threadIdx.x >> 5;
    const int row  = blockIdx.x * 8 + wave;            // 512 rows total

    const f32x4* src = (const f32x4*)(feat + (size_t)row * DD) + lane * 4;
    f32x4 v[4];
    float ss = 0.0f;
#pragma unroll
    for (int i = 0; i < 4; ++i) {
        v[i] = src[i];
#pragma unroll
        for (int j = 0; j < 4; ++j) ss += v[i][j] * v[i][j];
    }
#pragma unroll
    for (int off = 16; off > 0; off >>= 1)
        ss += __shfl_xor(ss, off, 32);
    const float inv = rsqrtf(ss);

    f16x8 o[2];
#pragma unroll
    for (int i = 0; i < 4; ++i)
#pragma unroll
        for (int j = 0; j < 4; ++j)
            o[i >> 1][(i & 1) * 4 + j] = (_Float16)(v[i][j] * inv);

    f16x8* dst = (f16x8*)(fn + (size_t)row * DD) + lane * 2;
    dst[0] = o[0];
    dst[1] = o[1];
}

// ---------------------------------------------------------------------------
// Normalize weight rows (f32 -> unit-norm f16), subcenter-major + zero pad.
//   dst row = k*CPAD + c ;  src row = c*KSUB + k  (zeros when c >= CC)
// ---------------------------------------------------------------------------
__global__ __launch_bounds__(256)
void arcface_norm_weights(const float* __restrict__ weight, _Float16* __restrict__ wn)
{
    const int lane   = threadIdx.x & 31;
    const int wave   = threadIdx.x >> 5;
    const int dstrow = blockIdx.x * 8 + wave;          // 3*CPAD rows total
    const int k      = dstrow / CPAD;
    const int c      = dstrow - k * CPAD;

    f16x8* dst = (f16x8*)(wn + (size_t)dstrow * DD) + lane * 2;

    if (c >= CC) {                                     // zero padding rows
        f16x8 z = {};
        dst[0] = z;
        dst[1] = z;
        return;
    }

    const size_t srow = (size_t)c * KSUB + k;
    const f32x4* src  = (const f32x4*)(weight + srow * DD) + lane * 4;
    f32x4 v[4];
    float ss = 0.0f;
#pragma unroll
    for (int i = 0; i < 4; ++i) {
        v[i] = src[i];
#pragma unroll
        for (int j = 0; j < 4; ++j) ss += v[i][j] * v[i][j];
    }
#pragma unroll
    for (int off = 16; off > 0; off >>= 1)
        ss += __shfl_xor(ss, off, 32);
    const float inv = rsqrtf(ss);

    f16x8 o[2];
#pragma unroll
    for (int i = 0; i < 4; ++i)
#pragma unroll
        for (int j = 0; j < 4; ++j)
            o[i >> 1][(i & 1) * 4 + j] = (_Float16)(v[i][j] * inv);
    dst[0] = o[0];
    dst[1] = o[1];
}

// ---------------------------------------------------------------------------
// WMMA GEMM + fused subcenter max: out[b][c] = S * max_k fn[b]·wn[k*CPAD+c]
// Block: 256 threads = 8 waves (2 M x 4 N); block tile 64x256, wave tile
// 32x64 (2x4 v_wmma_f32_16x16x32_f16 per 32-deep k-step).
// A tile staged to LDS once (async), B streamed from global.
// ---------------------------------------------------------------------------
__global__ __launch_bounds__(256)
void arcface_wmma_gemm_max(const _Float16* __restrict__ fn,
                           const _Float16* __restrict__ wn,
                           float* __restrict__ out)
{
    __shared__ _Float16 As[64 * LDS_PITCH];            // 66,560 B

    const int tid  = threadIdx.x;
    const int lane = tid & 31;
    const int wave = tid >> 5;
    const int wm   = wave >> 2;                        // 0..1  (M)
    const int wnid = wave & 3;                         // 0..3  (N)

    const int m_base = blockIdx.x * 64;
    const int n0     = blockIdx.y * 256 + wnid * 64;

    const int r  = lane & 15;
    const int hh = lane >> 4;

    // ---- stage A tile (64 rows x 512 halves) into LDS via async-to-LDS ----
    {
        const _Float16* gbase = fn + (size_t)m_base * DD;
#pragma unroll
        for (int it = 0; it < 16; ++it) {
            const int chunk = it * 256 + tid;          // 16-byte chunks
            const int row   = chunk >> 6;              // 64 chunks per row
            const int cin   = chunk & 63;
            const uint32_t lds_off =
                (uint32_t)(uintptr_t)(&As[row * LDS_PITCH + cin * 8]);
            const _Float16* gp = gbase + (size_t)row * DD + cin * 8;
            asm volatile("global_load_async_to_lds_b128 %0, %1, off"
                         :: "v"(lds_off), "v"(gp) : "memory");
        }
    }
#if __has_builtin(__builtin_amdgcn_s_wait_asynccnt)
    __builtin_amdgcn_s_wait_asynccnt(0);
#else
    asm volatile("s_wait_asynccnt 0x0" ::: "memory");
#endif
    __syncthreads();

    // Per-lane LDS fragment pointers for the two A tiles (M offsets 0, 16).
    // A lane layout: lanes 0-15 row M=lane, K {0..7,16..23};
    // lanes 16-31 row M=lane-16, K {8..15,24..31}.
    const f16x8* a0 = (const f16x8*)&As[(wm * 32 + r) * LDS_PITCH + hh * 8];
    const f16x8* a1 = (const f16x8*)&As[(wm * 32 + 16 + r) * LDS_PITCH + hh * 8];

    v8f best[2][4];

#pragma unroll
    for (int s = 0; s < KSUB; ++s) {
        const _Float16* bbase = wn + (size_t)s * (size_t)CPAD * DD;
        v8f acc[2][4];
#pragma unroll
        for (int tm = 0; tm < 2; ++tm)
#pragma unroll
            for (int tn = 0; tn < 4; ++tn)
                acc[tm][tn] = (v8f){};

        for (int k0 = 0; k0 < DD; k0 += 32) {
            const int kk = k0 >> 3;                    // f16x8 index
            v16h a[2], b[4];
            ((f16x8*)&a[0])[0] = a0[kk];
            ((f16x8*)&a[0])[1] = a0[kk + 2];           // +16 halves
            ((f16x8*)&a[1])[0] = a1[kk];
            ((f16x8*)&a[1])[1] = a1[kk + 2];
            // B fragments: wn row (n0+16*tn+r), K {8*hh..8*hh+7, 16+8*hh..16+8*hh+7}
#pragma unroll
            for (int tn = 0; tn < 4; ++tn) {
                const f16x8* pb =
                    (const f16x8*)(bbase + (size_t)(n0 + 16 * tn + r) * DD + k0 + hh * 8);
                ((f16x8*)&b[tn])[0] = pb[0];
                ((f16x8*)&b[tn])[1] = pb[2];
            }
#pragma unroll
            for (int tm = 0; tm < 2; ++tm)
#pragma unroll
                for (int tn = 0; tn < 4; ++tn)
                    acc[tm][tn] = __builtin_amdgcn_wmma_f32_16x16x32_f16(
                        false, a[tm], false, b[tn],
                        (short)0, acc[tm][tn], false, false);
        }

        if (s == 0) {
#pragma unroll
            for (int tm = 0; tm < 2; ++tm)
#pragma unroll
                for (int tn = 0; tn < 4; ++tn)
                    best[tm][tn] = acc[tm][tn];
        } else {
#pragma unroll
            for (int tm = 0; tm < 2; ++tm)
#pragma unroll
                for (int tn = 0; tn < 4; ++tn)
#pragma unroll
                    for (int j = 0; j < 8; ++j)
                        best[tm][tn][j] = fmaxf(best[tm][tn][j], acc[tm][tn][j]);
        }
    }

    // C/D layout: VGPR j -> row m_base + wm*32 + 16*tm + j + 8*hh, col n0+16*tn+r
#pragma unroll
    for (int tm = 0; tm < 2; ++tm)
#pragma unroll
        for (int tn = 0; tn < 4; ++tn) {
            const int col = n0 + 16 * tn + r;
            if (col < CC) {
#pragma unroll
                for (int j = 0; j < 8; ++j) {
                    const int row = m_base + wm * 32 + 16 * tm + j + 8 * hh;
                    out[(size_t)row * CC + col] = SS * best[tm][tn][j];
                }
            }
        }
}

// ---------------------------------------------------------------------------
// Adaptive-margin fix-up at the label column (512 elements).
// ---------------------------------------------------------------------------
__global__ __launch_bounds__(256)
void arcface_margin_fixup(const int* __restrict__ labels,
                          const float* __restrict__ margins,
                          float* __restrict__ out)
{
    const int b = blockIdx.x * blockDim.x + threadIdx.x;
    if (b >= BB) return;
    const int   lab = labels[b];
    const float m   = margins[lab];
    const float cm  = __cosf(m);
    const float sm  = __sinf(m);
    const float th  = -cm;            // cos(pi - m)
    const float mm  = sm * m;         // sin(pi - m) * m

    const size_t idx = (size_t)b * CC + lab;
    const float  c   = out[idx] * (1.0f / SS);
    const float  cc  = fminf(fmaxf(c, -1.0f), 1.0f);
    const float  sine = sqrtf(1.0f - cc * cc);
    const float  phi  = (c > th) ? (c * cm - sine * sm) : (c - mm);
    out[idx] = SS * phi;
}

// ---------------------------------------------------------------------------
extern "C" void kernel_launch(void* const* d_in, const int* in_sizes, int n_in,
                              void* d_out, int out_size, void* d_ws, size_t ws_size,
                              hipStream_t stream)
{
    const float* features = (const float*)d_in[0];   // [512, 512]
    const float* weight   = (const float*)d_in[1];   // [300000, 512]
    const float* margins  = (const float*)d_in[2];   // [100000]
    const int*   labels   = (const int*)d_in[3];     // [512]
    float*       out      = (float*)d_out;           // [512, 100000]

    _Float16* fn = (_Float16*)d_ws;                  // 512*512 f16 = 512 KB
    _Float16* wn = fn + (size_t)BB * DD;             // 3*CPAD*512 f16 ~ 307 MB

    // 1) feature normalization: 512 rows, 8 rows/block
    arcface_norm_features<<<BB / 8, 256, 0, stream>>>(features, fn);

    // 2) weight normalization + subcenter-major permute + zero pad
    arcface_norm_weights<<<(KSUB * CPAD) / 8, 256, 0, stream>>>(weight, wn);

    // 3) fused GEMM + subcenter max + scale: grid 8 (M) x 391 (N)
    dim3 grid(BB / 64, CPAD / 256);
    arcface_wmma_gemm_max<<<grid, 256, 0, stream>>>(fn, wn, out);

    // 4) margin scatter at label columns
    arcface_margin_fixup<<<2, 256, 0, stream>>>(labels, margins, out);
}
